// REGCN_26628797235282
// MI455X (gfx1250) — matrix operand order repaired and checked by reading, not compiled, not run on hardware
//
#include <hip/hip_runtime.h>
#include <math.h>

typedef __attribute__((ext_vector_type(16))) _Float16 v16h;
typedef __attribute__((ext_vector_type(4)))  _Float16 v4h;
typedef __attribute__((ext_vector_type(8)))  float    v8f;
typedef __attribute__((ext_vector_type(4)))  float    v4f;

#define GTB 256
static __device__ __forceinline__ float sigmoidf_(float x) { return 1.0f / (1.0f + expf(-x)); }
#define RRELU_SLOPE ((0.125f + 1.0f / 3.0f) * 0.5f)

// ---------------------------------------------------------------------------
// Pack W (Nout x K, row-major f32) into fragment-swizzled f16 B operand:
// frag (ntile,kstep) of 512 halves; half index = ((frag*32)+lane)*16 + i
// maps to B[k][n] = W[n*K + k] with n = ntile*16 + (lane&15),
//                               k = kstep*32 + ((lane>>4)<<4) + i.
// ---------------------------------------------------------------------------
__global__ void k_pack_b(_Float16* __restrict__ dst, const float* __restrict__ W,
                         int Nout, int Kdim, int ksteps, int ntiles) {
  int idx = blockIdx.x * blockDim.x + threadIdx.x;
  int total = ntiles * ksteps * 512;
  if (idx >= total) return;
  int i    = idx & 15;
  int lane = (idx >> 4) & 31;
  int frag = idx >> 9;
  int ks = frag % ksteps;
  int nt = frag / ksteps;
  int n = nt * 16 + (lane & 15);
  int k = ks * 32 + ((lane >> 4) << 4) + i;
  float v = (n < Nout && k < Kdim) ? W[(size_t)n * Kdim + k] : 0.0f;
  dst[idx] = (_Float16)v;
}

// ---------------------------------------------------------------------------
// Generic WMMA GEMM: C(MxNout) = A(MxK) * B(KxNout)  (B pre-packed f16 swz)
//  - optional row gather idxA, optional second gathered addend A2/idxA2
//  - output: direct store (+bias,+relu) OR atomic scatter rows via idxOut
//  Block: 256 thr (8 waves). Each wave owns a 32x32 output tile
//  (2 row-tiles x 2 n-tiles -> 4 WMMAs per K-step). A staged through LDS in
//  32x32 chunks: each thread owns one (row, 4k) slice -> one global_load_b128
//  + one ds_store_b64 per chunk; gather indices live in registers.
// ---------------------------------------------------------------------------
__global__ __launch_bounds__(GTB) void k_wmma_gemm(
    const float* __restrict__ A, int ldA, const int* __restrict__ idxA,
    const float* __restrict__ A2, int ldA2, const int* __restrict__ idxA2,
    const _Float16* __restrict__ Bswz,
    int M, int K, int Nout, int Kpad, int Npad,
    float* __restrict__ Cout, int ldC,
    const float* __restrict__ bias, int act,
    const int* __restrict__ idxOut) {
  __shared__ _Float16 As[32 * 32];

  const int lane = threadIdx.x & 31;
  const int wave = threadIdx.x >> 5;
  const int m0 = blockIdx.x * 32;
  const int ksteps = Kpad >> 5;
  const int ntiles = Npad >> 4;
  const int ntile0 = (blockIdx.y * 8 + wave) * 2;
  const bool act0 = (ntile0 < ntiles);
  const bool act1 = (ntile0 + 1 < ntiles);

  // --- staging assignment: this thread stages row sr, k in [sk0, sk0+4) ---
  const int sr = threadIdx.x >> 3;         // 0..31
  const int sk0 = (threadIdx.x & 7) << 2;  // 0,4,...,28
  int riS = -1, ri2S = -1;
  {
    int m = m0 + sr;
    if (m < M) {
      riS = idxA ? idxA[m] : m;
      if (idxA2) ri2S = idxA2[m];
    }
  }

  v8f acc00 = {};  // rowtile0 x ntile0
  v8f acc10 = {};  // rowtile1 x ntile0
  v8f acc01 = {};  // rowtile0 x ntile1
  v8f acc11 = {};  // rowtile1 x ntile1
  const int mA = lane & 15;
  const int kbase = (lane >> 4) << 4;  // 0 or 16

  const _Float16* bptr0 = Bswz + (((size_t)ntile0 * ksteps) * 32 + lane) * 16;
  const _Float16* bptr1 = bptr0 + (size_t)ksteps * 512;

  for (int ks = 0; ks < ksteps; ++ks) {
    __syncthreads();
    // stage A chunk: 32 rows x 32 k (gather + add + f32->f16), vectorized
    {
      int k = ks * 32 + sk0;
      v4f av = {0.0f, 0.0f, 0.0f, 0.0f};
      if (riS >= 0 && k < K) {
        if (k + 4 <= K) {
          av = *(const v4f*)(A + (size_t)riS * ldA + k);
          if (A2) av += *(const v4f*)(A2 + (size_t)ri2S * ldA2 + k);
        } else {
          const float* ap = A + (size_t)riS * ldA;
          const float* ap2 = A2 ? (A2 + (size_t)ri2S * ldA2) : (const float*)0;
#pragma unroll
          for (int q = 0; q < 4; ++q)
            if (k + q < K) {
              av[q] = ap[k + q];
              if (ap2) av[q] += ap2[k + q];
            }
        }
      }
      v4h hv;
#pragma unroll
      for (int q = 0; q < 4; ++q) hv[q] = (_Float16)av[q];
      *(v4h*)&As[sr * 32 + sk0] = hv;
    }
    __syncthreads();
    if (act0) {
      v16h a0, a1;
      const _Float16* ap0 = &As[mA * 32 + kbase];
      const _Float16* ap1 = &As[(mA + 16) * 32 + kbase];
#pragma unroll
      for (int i = 0; i < 8; ++i) {
        a0[i] = ap0[i];      a0[8 + i] = ap0[8 + i];
        a1[i] = ap1[i];      a1[8 + i] = ap1[8 + i];
      }
      v16h b0 = *(const v16h*)(bptr0 + (size_t)ks * 512);
      if (ks + 1 < ksteps) __builtin_prefetch(bptr0 + (size_t)(ks + 1) * 512, 0, 1);
      acc00 = __builtin_amdgcn_wmma_f32_16x16x32_f16(false, a0, false, b0, (short)0, acc00, false, false);
      acc10 = __builtin_amdgcn_wmma_f32_16x16x32_f16(false, a1, false, b0, (short)0, acc10, false, false);
      if (act1) {
        v16h b1 = *(const v16h*)(bptr1 + (size_t)ks * 512);
        if (ks + 1 < ksteps) __builtin_prefetch(bptr1 + (size_t)(ks + 1) * 512, 0, 1);
        acc01 = __builtin_amdgcn_wmma_f32_16x16x32_f16(false, a0, false, b1, (short)0, acc01, false, false);
        acc11 = __builtin_amdgcn_wmma_f32_16x16x32_f16(false, a1, false, b1, (short)0, acc11, false, false);
      }
    }
  }
  if (!act0) return;

  const int mhalf = (lane >> 4) << 3;  // 0 or 8
#pragma unroll
  for (int s = 0; s < 2; ++s) {
    if (s == 1 && !act1) break;
    int n = (ntile0 + s) * 16 + (lane & 15);
    if (n >= Nout) continue;
    v8f accA = s ? acc01 : acc00;
    v8f accB = s ? acc11 : acc10;
#pragma unroll
    for (int r = 0; r < 8; ++r) {
      int m = m0 + r + mhalf;  // row tile 0
      if (m < M) {
        float v = accA[r];
        if (idxOut) {
          atomicAdd(&Cout[(size_t)idxOut[m] * ldC + n], v);
        } else {
          if (bias) v += bias[n];
          if (act == 1) v = fmaxf(v, 0.0f);
          Cout[(size_t)m * ldC + n] = v;
        }
      }
      int m2 = m0 + 16 + r + mhalf;  // row tile 1
      if (m2 < M) {
        float v = accB[r];
        if (idxOut) {
          atomicAdd(&Cout[(size_t)idxOut[m2] * ldC + n], v);
        } else {
          if (bias) v += bias[n];
          if (act == 1) v = fmaxf(v, 0.0f);
          Cout[(size_t)m2 * ldC + n] = v;
        }
      }
    }
  }
}

// ---------------------------------------------------------------------------
// Elementwise / reduction helpers
// ---------------------------------------------------------------------------
__global__ void k_l2norm(float* __restrict__ dst, const float* __restrict__ src, int H) {
  int r = blockIdx.x;
  __shared__ float red[GTB];
  float s = 0.0f;
  for (int j = threadIdx.x; j < H; j += blockDim.x) {
    float v = src[(size_t)r * H + j];
    s += v * v;
  }
  red[threadIdx.x] = s;
  __syncthreads();
  for (int off = GTB / 2; off > 0; off >>= 1) {
    if (threadIdx.x < off) red[threadIdx.x] += red[threadIdx.x + off];
    __syncthreads();
  }
  float inv = 1.0f / fmaxf(sqrtf(red[0]), 1e-12f);
  for (int j = threadIdx.x; j < H; j += blockDim.x)
    dst[(size_t)r * H + j] = src[(size_t)r * H + j] * inv;
}

__global__ void k_count(float* __restrict__ cnt, const int* __restrict__ idx, int n) {
  int i = blockIdx.x * blockDim.x + threadIdx.x;
  if (i < n) atomicAdd(&cnt[idx[i]], 1.0f);
}

__global__ void k_scatter_rows(float* __restrict__ sums, const float* __restrict__ src,
                               const int* __restrict__ rowIdx, const int* __restrict__ segIdx,
                               int total, int H) {
  int i = blockIdx.x * blockDim.x + threadIdx.x;
  if (i >= total) return;
  int e = i / H;
  int k = i - e * H;
  atomicAdd(&sums[(size_t)segIdx[e] * H + k], src[(size_t)rowIdx[e] * H + k]);
}

__global__ void k_rel_ent(float* __restrict__ out, const float* __restrict__ sums,
                          const float* __restrict__ cnt, int total, int H) {
  int i = blockIdx.x * blockDim.x + threadIdx.x;
  if (i >= total) return;
  float c = cnt[i / H];
  out[i] = (c > 0.0f) ? sums[i] / fmaxf(c, 1.0f) : 0.0f;
}

__global__ void k_concat(float* __restrict__ x, const float* __restrict__ a,
                         const float* __restrict__ b, int R, int H) {
  int i = blockIdx.x * blockDim.x + threadIdx.x;
  int total = R * 2 * H;
  if (i >= total) return;
  int r = i / (2 * H);
  int j = i - r * 2 * H;
  x[i] = (j < H) ? a[(size_t)r * H + j] : b[(size_t)r * H + (j - H)];
}

__global__ void k_gru(float* __restrict__ newh, const float* __restrict__ gi,
                      const float* __restrict__ gh, const float* __restrict__ h,
                      int R, int H) {
  int i = blockIdx.x * blockDim.x + threadIdx.x;
  if (i >= R * H) return;
  int r = i / H;
  int j = i - r * H;
  const float* gir = gi + (size_t)r * 3 * H;
  const float* ghr = gh + (size_t)r * 3 * H;
  float rr = sigmoidf_(gir[j] + ghr[j]);
  float z  = sigmoidf_(gir[H + j] + ghr[H + j]);
  float nn = tanhf(gir[2 * H + j] + rr * ghr[2 * H + j]);
  newh[i] = (1.0f - z) * nn + z * h[i];
}

__global__ void k_combine(float* __restrict__ hout, const float* __restrict__ agg,
                          const float* __restrict__ deg, const float* __restrict__ S,
                          const float* __restrict__ hin, const float* __restrict__ Wel,
                          int N, int H) {
  int i = blockIdx.x * blockDim.x + threadIdx.x;
  if (i >= N * H) return;
  int v = i / H;
  int j = i - v * H;
  float d = deg[v];
  float val;
  if (d > 0.0f) {
    val = agg[i] / d + S[i];
  } else {  // isolated node: self message through Wel
    float s = 0.0f;
    for (int k = 0; k < H; ++k) s += hin[(size_t)v * H + k] * Wel[(size_t)j * H + k];
    val = s;
  }
  hout[i] = (val >= 0.0f) ? val : RRELU_SLOPE * val;
}

__global__ void k_ent_update(float* __restrict__ ent, const float* __restrict__ uPre,
                             const float* __restrict__ went, int total) {
  int i = blockIdx.x * blockDim.x + threadIdx.x;
  if (i >= total) return;
  float u = sigmoidf_(uPre[i]);
  ent[i] = ent[i] + u * (went[i] - ent[i]);
}

__global__ void k_conv(float* __restrict__ out, const float* __restrict__ X1,
                       const int* __restrict__ i1, const float* __restrict__ X2,
                       const int* __restrict__ i2, const float* __restrict__ cw,
                       const float* __restrict__ cb, int Bn, int H, int C, int Kk) {
  int idx = blockIdx.x * blockDim.x + threadIdx.x;
  int total = Bn * C * H;
  if (idx >= total) return;
  int pos = idx % H;
  int c = (idx / H) % C;
  int b = idx / (H * C);
  const float* r1 = X1 + (size_t)i1[b] * H;
  const float* r2 = X2 + (size_t)i2[b] * H;
  float s = cb[c];
  for (int kk = 0; kk < Kk; ++kk) {
    int p = pos + kk - Kk / 2;
    if (p >= 0 && p < H) {
      s += r1[p] * cw[c * 2 * Kk + kk];
      s += r2[p] * cw[c * 2 * Kk + Kk + kk];
    }
  }
  out[(size_t)b * (C * H) + c * H + pos] = fmaxf(s, 0.0f);
}

// ---------------------------------------------------------------------------
extern "C" void kernel_launch(void* const* d_in, const int* in_sizes, int n_in,
                              void* d_out, int out_size, void* d_ws, size_t ws_size,
                              hipStream_t stream) {
  (void)in_sizes; (void)n_in; (void)out_size; (void)ws_size;
  const int N = 20000, R = 400, H = 200, E = 200000, T = 3, Bb = 1024, C = 50, Kk = 3;

  const float* ent_embeds = (const float*)d_in[0];
  const float* rel_embeds = (const float*)d_in[1];
  const float* Wr1 = (const float*)d_in[2];
  const float* Wsl1 = (const float*)d_in[3];
  const float* Wel1 = (const float*)d_in[4];
  const float* Wr2 = (const float*)d_in[5];
  const float* Wsl2 = (const float*)d_in[6];
  const float* Wel2 = (const float*)d_in[7];
  const float* lin_W = (const float*)d_in[8];
  const float* lin_b = (const float*)d_in[9];
  const float* gWih = (const float*)d_in[10];
  const float* gWhh = (const float*)d_in[11];
  const float* gbih = (const float*)d_in[12];
  const float* gbhh = (const float*)d_in[13];
  const float* conve_w = (const float*)d_in[14];
  const float* conve_b = (const float*)d_in[15];
  const float* fce_w = (const float*)d_in[16];
  const float* fce_b = (const float*)d_in[17];
  const float* convr_w = (const float*)d_in[18];
  const float* convr_b = (const float*)d_in[19];
  const float* fcr_w = (const float*)d_in[20];
  const float* fcr_b = (const float*)d_in[21];
  const int* src = (const int*)d_in[22];
  const int* dst = (const int*)d_in[23];
  const int* erel = (const int*)d_in[24];
  const int* subj = (const int*)d_in[25];
  const int* rel = (const int*)d_in[26];
  const int* obj = (const int*)d_in[27];
  float* out = (float*)d_out;

  char* ws = (char*)d_ws;
  size_t off = 0;
  auto alloc = [&](size_t bytes) -> void* {
    off = (off + 255) & ~(size_t)255;
    void* p = ws + off;
    off += bytes;
    return p;
  };
  auto allocF = [&](size_t n) -> float* { return (float*)alloc(n * sizeof(float)); };

  float* ent = allocF((size_t)N * H);
  float* relA = allocF((size_t)R * H);
  float* relB = allocF((size_t)R * H);
  float* sums = allocF((size_t)R * H);
  float* rel_ent = allocF((size_t)R * H);
  float* xcat = allocF((size_t)R * 2 * H);
  float* gi = allocF((size_t)R * 3 * H);
  float* gh = allocF((size_t)R * 3 * H);
  float* newh = allocF((size_t)R * H);
  float* cnt = allocF(R);
  float* deg = allocF(N);
  float* agg = allocF((size_t)N * H);   // also reused as uPre
  float* Sm = allocF((size_t)N * H);
  float* h1 = allocF((size_t)N * H);    // also reused as w_ent
  float* h2 = allocF((size_t)N * H);
  float* featE = allocF((size_t)Bb * H);
  float* featR = allocF((size_t)Bb * H);
  float* convbuf = allocF((size_t)Bb * C * H);

  auto packB = [&](const float* W, int Nout, int Kdim) -> _Float16* {
    int ksteps = (Kdim + 31) / 32;
    int ntiles = (Nout + 15) / 16;
    size_t sz = (size_t)ntiles * ksteps * 512;
    _Float16* p = (_Float16*)alloc(sz * sizeof(_Float16));
    int total = (int)sz;
    k_pack_b<<<(total + GTB - 1) / GTB, GTB, 0, stream>>>(p, W, Nout, Kdim, ksteps, ntiles);
    return p;
  };

  auto gemm = [&](const float* A, int ldA, const int* idxA, const float* A2, int ldA2,
                  const int* idxA2, const _Float16* Bm, int M, int Kdim, int Nout,
                  float* Cm, int ldC, const float* bias, int act, const int* idxOut) {
    int Kpad = ((Kdim + 31) / 32) * 32;
    int Npad = ((Nout + 15) / 16) * 16;
    int ntiles = Npad / 16;
    dim3 g((M + 31) / 32, (ntiles + 15) / 16);  // each wave: 2 n-tiles, 8 waves/block
    k_wmma_gemm<<<g, GTB, 0, stream>>>(A, ldA, idxA, A2, ldA2, idxA2, Bm, M, Kdim, Nout,
                                       Kpad, Npad, Cm, ldC, bias, act, idxOut);
  };

  // ---- pack all static weights to f16 fragment layout (once per call) ----
  _Float16* pkWr1 = packB(Wr1, H, H);
  _Float16* pkWsl1 = packB(Wsl1, H, H);
  _Float16* pkWr2 = packB(Wr2, H, H);
  _Float16* pkWsl2 = packB(Wsl2, H, H);
  _Float16* pkLin = packB(lin_W, H, H);
  _Float16* pkWih = packB(gWih, 3 * H, 2 * H);
  _Float16* pkWhh = packB(gWhh, 3 * H, H);
  _Float16* pkFce = packB(fce_w, H, C * H);
  _Float16* pkFcr = packB(fcr_w, H, C * H);

  // ---- initial normalization ----
  k_l2norm<<<N, GTB, 0, stream>>>(ent, ent_embeds, H);
  float* rel_h = relA;
  float* nrel = relB;
  k_l2norm<<<R, GTB, 0, stream>>>(rel_h, rel_embeds, H);

  for (int t = 0; t < T; ++t) {
    const int* src_t = src + (size_t)t * E;
    const int* dst_t = dst + (size_t)t * E;
    const int* erel_t = erel + (size_t)t * E;

    hipMemsetAsync(sums, 0, (size_t)R * H * sizeof(float), stream);
    hipMemsetAsync(cnt, 0, (size_t)R * sizeof(float), stream);
    hipMemsetAsync(deg, 0, (size_t)N * sizeof(float), stream);
    k_count<<<(E + GTB - 1) / GTB, GTB, 0, stream>>>(cnt, erel_t, E);
    k_count<<<(E + GTB - 1) / GTB, GTB, 0, stream>>>(deg, dst_t, E);
    k_scatter_rows<<<(E * H + GTB - 1) / GTB, GTB, 0, stream>>>(sums, ent, src_t, erel_t, E * H, H);
    k_rel_ent<<<(R * H + GTB - 1) / GTB, GTB, 0, stream>>>(rel_ent, sums, cnt, R * H, H);
    k_concat<<<(R * 2 * H + GTB - 1) / GTB, GTB, 0, stream>>>(xcat, rel_embeds, rel_ent, R, H);

    // GRU gates: gi = x @ Wih^T + bih ; gh = rel_h @ Whh^T + bhh
    gemm(xcat, 2 * H, nullptr, nullptr, 0, nullptr, pkWih, R, 2 * H, 3 * H, gi, 3 * H, gbih, 0, nullptr);
    gemm(rel_h, H, nullptr, nullptr, 0, nullptr, pkWhh, R, H, 3 * H, gh, 3 * H, gbhh, 0, nullptr);
    k_gru<<<(R * H + GTB - 1) / GTB, GTB, 0, stream>>>(newh, gi, gh, rel_h, R, H);
    k_l2norm<<<R, GTB, 0, stream>>>(nrel, newh, H);

    // RGCN layer 1: fused gather + WMMA + atomic scatter
    hipMemsetAsync(agg, 0, (size_t)N * H * sizeof(float), stream);
    gemm(ent, H, src_t, nrel, H, erel_t, pkWr1, E, H, H, agg, H, nullptr, 0, dst_t);
    gemm(ent, H, nullptr, nullptr, 0, nullptr, pkWsl1, N, H, H, Sm, H, nullptr, 0, nullptr);
    k_combine<<<(N * H + GTB - 1) / GTB, GTB, 0, stream>>>(h1, agg, deg, Sm, ent, Wel1, N, H);

    // RGCN layer 2
    hipMemsetAsync(agg, 0, (size_t)N * H * sizeof(float), stream);
    gemm(h1, H, src_t, nrel, H, erel_t, pkWr2, E, H, H, agg, H, nullptr, 0, dst_t);
    gemm(h1, H, nullptr, nullptr, 0, nullptr, pkWsl2, N, H, H, Sm, H, nullptr, 0, nullptr);
    k_combine<<<(N * H + GTB - 1) / GTB, GTB, 0, stream>>>(h2, agg, deg, Sm, h1, Wel2, N, H);

    // w_ent = l2norm(h2) (into h1); u = sigmoid(ent@lin_W^T + b) (pre-act into agg)
    k_l2norm<<<N, GTB, 0, stream>>>(h1, h2, H);
    gemm(ent, H, nullptr, nullptr, 0, nullptr, pkLin, N, H, H, agg, H, lin_b, 0, nullptr);
    k_ent_update<<<(N * H + GTB - 1) / GTB, GTB, 0, stream>>>(ent, agg, h1, N * H);

    float* tmp = rel_h; rel_h = nrel; nrel = tmp;  // rel_h <- n_rel
  }

  // ---- decoder ----
  _Float16* pkEntT = packB(ent, N, H);    // B[k][n] = ent[n][k]
  _Float16* pkRelT = packB(rel_h, R, H);  // B[k][n] = rel_h[n][k]

  // obj path: conv(ent[subj], rel_h[rel]) -> fc -> @ ent^T
  k_conv<<<(Bb * C * H + GTB - 1) / GTB, GTB, 0, stream>>>(convbuf, ent, subj, rel_h, rel,
                                                           conve_w, conve_b, Bb, H, C, Kk);
  gemm(convbuf, C * H, nullptr, nullptr, 0, nullptr, pkFce, Bb, C * H, H, featE, H, fce_b, 1, nullptr);
  gemm(featE, H, nullptr, nullptr, 0, nullptr, pkEntT, Bb, H, N, out, N, nullptr, 0, nullptr);

  // rel path: conv(ent[subj], ent[obj]) -> fc -> @ rel_h^T
  k_conv<<<(Bb * C * H + GTB - 1) / GTB, GTB, 0, stream>>>(convbuf, ent, subj, ent, obj,
                                                           convr_w, convr_b, Bb, H, C, Kk);
  gemm(convbuf, C * H, nullptr, nullptr, 0, nullptr, pkFcr, Bb, C * H, H, featR, H, fcr_b, 1, nullptr);
  gemm(featR, H, nullptr, nullptr, 0, nullptr, pkRelT, Bb, H, R, out + (size_t)Bb * N, R, nullptr, 0, nullptr);
}
